// SCAREDecoder_58213986730285
// MI455X (gfx1250) — matrix operand order repaired and checked by reading, not compiled, not run on hardware
//
#include <hip/hip_runtime.h>
#include <math.h>

// ---------------- model constants ----------------
#define LAY 4
#define Hd  768
#define NHd 12
#define DHd 64
#define FFd 3072
#define Bd  4
#define Sd  1024
#define Md  (Bd * Sd)   // 4096 token rows

typedef __attribute__((ext_vector_type(16))) __bf16 v16bf;
typedef __attribute__((ext_vector_type(8)))  float  v8f;
typedef __attribute__((ext_vector_type(4)))  unsigned int v4u;
typedef __attribute__((ext_vector_type(8)))  int    v8i;
typedef __attribute__((ext_vector_type(4)))  int    v4i;

// ---------------- helpers ----------------
__device__ __forceinline__ unsigned short f2bf(float x) {
    unsigned u = __float_as_uint(x);
    unsigned r = u + 0x7FFFu + ((u >> 16) & 1u);
    return (unsigned short)(r >> 16);
}

__device__ __forceinline__ v8f wmma_bf16(v16bf a, v16bf b, v8f c) {
    return __builtin_amdgcn_wmma_f32_16x16x32_bf16(false, a, false, b, (short)0, c, false, false);
}

// low 32 bits of a generic pointer into LDS == LDS byte address
__device__ __forceinline__ unsigned lds_lo32(const void* p) {
    return (unsigned)(unsigned long long)(uintptr_t)p;
}

// Tensor Data Mover: DMA a 2D tile (tile_d0 x tile_d1 of 2-byte elements, row stride
// stride_d0 elements) from global memory to LDS.  D# layout per CDNA5 ISA.
// This toolchain exposes the 6-arg builtin: (v4u g0, v8i g1, v4i g2, v4i g3, v8i extra, i32 cpol).
__device__ __forceinline__ void tdm_load_2d(unsigned lds_addr, const void* gptr,
                                            unsigned tile_d0, unsigned tile_d1,
                                            unsigned stride_d0) {
    unsigned long long ga = (unsigned long long)(uintptr_t)gptr;
    v4u g0;
    g0.x = 1u;                                        // count=1, user descriptor
    g0.y = lds_addr;                                  // lds_addr (bytes)
    g0.z = (unsigned)(ga & 0xFFFFFFFFu);              // global_addr[31:0]
    g0.w = (unsigned)((ga >> 32) & 0x01FFFFFFu) | (2u << 30); // global_addr[56:32] | type=2
    v8i g1;
    g1[0] = (int)(1u << 16);                          // workgroup_mask=0, data_size=1 (2B)
    g1[1] = (int)((tile_d0 & 0xFFFFu) << 16);         // abar_addr=0 | tensor_dim0[15:0]=tile_d0
    g1[2] = (int)(((tile_d0 >> 16) & 0xFFFFu) | ((tile_d1 & 0xFFFFu) << 16)); // dim0 hi | dim1 lo
    g1[3] = (int)(((tile_d1 >> 16) & 0xFFFFu) | ((tile_d0 & 0xFFFFu) << 16)); // dim1 hi | tile_dim0
    g1[4] = (int)(tile_d1 & 0xFFFFu);                 // tile_dim1 | tile_dim2=0
    g1[5] = (int)stride_d0;                           // tensor_dim0_stride[31:0] (elements)
    g1[6] = 0;                                        // stride hi | dim1_stride lo (unused, 2D)
    g1[7] = 0;
    v4i gz4 = {0, 0, 0, 0};
    v8i gz8 = {0, 0, 0, 0, 0, 0, 0, 0};
    __builtin_amdgcn_tensor_load_to_lds(g0, g1, gz4, gz4, gz8, 0);
}

// A-fragment: 16x32 bf16, rows row0..row0+15, cols k0..k0+31, row-major src, stride ld
// lane L: m = row0 + (L&15), h = L>>4 ; element e <-> k = k0 + (e&7) + 8h + 16*(e>>3)
__device__ __forceinline__ v16bf frag_ld_a(const unsigned short* base, int ld, int row0, int k0, int lane) {
    int m = row0 + (lane & 15);
    int h = (lane >> 4) & 1;
    const unsigned short* p = base + (size_t)m * ld + k0 + 8 * h;
    union { v16bf v; uint4 q[2]; } f;
    f.q[0] = *reinterpret_cast<const uint4*>(p);
    f.q[1] = *reinterpret_cast<const uint4*>(p + 16);
    return f.v;
}

// B-fragment: 32x16 bf16 (K x N), source holds B^T row-major: row n, consecutive k, stride ld
// lane L: n = n0 + (L&15), h = L>>4 ; element e <-> k = k0 + (e&7) + 8h + 16*(e>>3)
__device__ __forceinline__ v16bf frag_ld_b(const unsigned short* base, int ld, int n0, int k0, int lane) {
    int n = n0 + (lane & 15);
    int h = (lane >> 4) & 1;
    const unsigned short* p = base + (size_t)n * ld + k0 + 8 * h;
    union { v16bf v; uint4 q[2]; } f;
    f.q[0] = *reinterpret_cast<const uint4*>(p);
    f.q[1] = *reinterpret_cast<const uint4*>(p + 16);
    return f.v;
}

// ---------------- prep kernels ----------------
// dst[l][n][k] = bf16(src[l][k][n])
__global__ void k_w_transpose(const float* __restrict__ src, unsigned short* __restrict__ dst,
                              int K, int N, int total) {
    int tid = blockIdx.x * blockDim.x + threadIdx.x;
    if (tid >= total) return;
    int l = tid / (K * N);
    int rem = tid - l * (K * N);
    int k = rem / N;
    int n = rem - k * N;
    dst[(size_t)l * K * N + (size_t)n * K + k] = f2bf(src[tid]);
}

// out_f32 = in ; out_bf = bf16(in)
__global__ void k_init(const float* __restrict__ in, float* __restrict__ h,
                       unsigned short* __restrict__ hbf, int total) {
    int tid = blockIdx.x * blockDim.x + threadIdx.x;
    if (tid >= total) return;
    float v = in[tid];
    h[tid] = v;
    hbf[tid] = f2bf(v);
}

// per (b,s): RoPE sin/cos tables (32 pair angles each for sc1/sc2), bbox coords, mask bias
__global__ void k_tables(const int* __restrict__ bbox, const int* __restrict__ amask,
                         float* __restrict__ s1, float* __restrict__ c1,
                         float* __restrict__ s2, float* __restrict__ c2,
                         float4* __restrict__ coords, float* __restrict__ extb) {
    int tid = blockIdx.x * blockDim.x + threadIdx.x;
    if (tid >= Bd * Sd) return;
    const int* bb = bbox + (size_t)tid * 4;
    float x1 = (float)bb[0], y1 = (float)bb[1], x2 = (float)bb[2], y2 = (float)bb[3];
    coords[tid] = make_float4(x1, x2, 0.5f * (x1 + x2), 0.5f * (y1 + y2));
    extb[tid] = (1.0f - (float)amask[tid]) * -10000.0f;
    size_t base = (size_t)tid * 32;
#pragma unroll
    for (int j = 0; j < 16; ++j) {
        float f = __expf(-(float)j * (9.210340371976184f / 16.0f)); // 10000^(-j/16)
        float ax1 = x1 * f, ay1 = y1 * f, ax2 = x2 * f, ay2 = y2 * f;
        s1[base + 2 * j]     = sinf(ax1);  c1[base + 2 * j]     = cosf(ax1);
        s1[base + 2 * j + 1] = sinf(ay1);  c1[base + 2 * j + 1] = cosf(ay1);
        s2[base + 2 * j]     = sinf(ax2);  c2[base + 2 * j]     = cosf(ax2);
        s2[base + 2 * j + 1] = sinf(ay2);  c2[base + 2 * j + 1] = cosf(ay2);
    }
}

// RoPE apply: C[M,H] f32 -> outbf[b,head,s,d] bf16 (scale folded in)
__global__ void k_rope(const float* __restrict__ C, unsigned short* __restrict__ outbf,
                       const float* __restrict__ s1, const float* __restrict__ c1,
                       const float* __restrict__ s2, const float* __restrict__ c2,
                       float scale) {
    int tid = blockIdx.x * blockDim.x + threadIdx.x;
    if (tid >= Bd * Sd * NHd * 32) return;
    int m = tid & 31;
    int t = tid >> 5;
    int head = t % NHd; t /= NHd;
    int s = t % Sd; int b = t / Sd;
    size_t inoff = ((size_t)(b * Sd + s)) * Hd + head * DHd + 2 * m;
    float x0 = C[inoff], x1v = C[inoff + 1];
    size_t tb = (size_t)(b * Sd + s) * 32 + m;
    float sn, cs;
    if ((head & 1) == 0) { sn = s1[tb]; cs = c1[tb]; }
    else                 { sn = s2[tb]; cs = c2[tb]; }
    float o0 = (x0 * cs - x1v * sn) * scale;
    float o1 = (x1v * cs + x0 * sn) * scale;
    size_t outoff = (((size_t)b * NHd + head) * Sd + s) * DHd + 2 * m;
    outbf[outoff] = f2bf(o0);
    outbf[outoff + 1] = f2bf(o1);
}

// V reorder: vC[M,H] f32 -> vt[b,head,d,s] bf16
__global__ void k_vre(const float* __restrict__ vC, unsigned short* __restrict__ vt) {
    int tid = blockIdx.x * blockDim.x + threadIdx.x;
    if (tid >= Bd * NHd * DHd * Sd) return;
    int s = tid % Sd; int t = tid / Sd;
    int d = t % DHd; t /= DHd;
    int head = t % NHd; int b = t / NHd;
    float v = vC[((size_t)b * Sd + s) * Hd + head * DHd + d];
    vt[(((size_t)b * NHd + head) * DHd + d) * Sd + s] = f2bf(v);
}

// ---------------- GEMM: C[M,N] = A_bf16[M,K] * Bt_bf16[N,K]^T + bias ; op=1 -> GELU ----------------
// Block tile 128x128, K-step 32.  A/B tiles double-buffered in LDS, filled by the
// Tensor Data Mover (wave 0 issues tensor_load_to_lds; all waves sync on TENSORcnt + barrier).
__global__ __launch_bounds__(256) void k_gemm_bf16(const unsigned short* __restrict__ A,
                                                   const unsigned short* __restrict__ Bt,
                                                   const float* __restrict__ bias,
                                                   float* __restrict__ Cf,
                                                   unsigned short* __restrict__ Cbf,
                                                   int Nd_, int Kd_, int op) {
    __shared__ unsigned short Als[2][128 * 32];
    __shared__ unsigned short Bls[2][128 * 32];

    const int lane = threadIdx.x & 31;
    const int wave = threadIdx.x >> 5;
    const int h = (lane >> 4) & 1;
    const int ln = lane & 15;
    const int rowBlk = blockIdx.x * 128;
    const int colBlk = blockIdx.y * 128;
    const int row0 = (wave >> 1) * 32;   // wave tile: 32 rows x 64 cols within the 128x128 block
    const int col0 = (wave & 1) * 64;

    // prologue: DMA buffer 0
    if (wave == 0) {
        tdm_load_2d(lds_lo32(&Als[0][0]), A + (size_t)rowBlk * Kd_, 32, 128, (unsigned)Kd_);
        tdm_load_2d(lds_lo32(&Bls[0][0]), Bt + (size_t)colBlk * Kd_, 32, 128, (unsigned)Kd_);
    }

    const v8f zf = {0.f, 0.f, 0.f, 0.f, 0.f, 0.f, 0.f, 0.f};
    v8f acc[2][4];
#pragma unroll
    for (int mi = 0; mi < 2; ++mi)
#pragma unroll
        for (int ni = 0; ni < 4; ++ni) acc[mi][ni] = zf;

    int cur = 0;
    for (int k0 = 0; k0 < Kd_; k0 += 32) {
        const bool hasNext = (k0 + 32 < Kd_);
        if (wave == 0 && hasNext) {
            tdm_load_2d(lds_lo32(&Als[cur ^ 1][0]), A + (size_t)rowBlk * Kd_ + k0 + 32, 32, 128, (unsigned)Kd_);
            tdm_load_2d(lds_lo32(&Bls[cur ^ 1][0]), Bt + (size_t)colBlk * Kd_ + k0 + 32, 32, 128, (unsigned)Kd_);
        }
        if (hasNext) __builtin_amdgcn_s_wait_tensorcnt(2);
        else         __builtin_amdgcn_s_wait_tensorcnt(0);
        __syncthreads();

        const unsigned short* Ab = &Als[cur][0];
        const unsigned short* Bb = &Bls[cur][0];
        v16bf a0 = frag_ld_a(Ab, 32, row0, 0, lane);
        v16bf a1 = frag_ld_a(Ab, 32, row0 + 16, 0, lane);
        v16bf b0 = frag_ld_b(Bb, 32, col0, 0, lane);
        v16bf b1 = frag_ld_b(Bb, 32, col0 + 16, 0, lane);
        v16bf b2 = frag_ld_b(Bb, 32, col0 + 32, 0, lane);
        v16bf b3 = frag_ld_b(Bb, 32, col0 + 48, 0, lane);
        acc[0][0] = wmma_bf16(a0, b0, acc[0][0]);
        acc[0][1] = wmma_bf16(a0, b1, acc[0][1]);
        acc[0][2] = wmma_bf16(a0, b2, acc[0][2]);
        acc[0][3] = wmma_bf16(a0, b3, acc[0][3]);
        acc[1][0] = wmma_bf16(a1, b0, acc[1][0]);
        acc[1][1] = wmma_bf16(a1, b1, acc[1][1]);
        acc[1][2] = wmma_bf16(a1, b2, acc[1][2]);
        acc[1][3] = wmma_bf16(a1, b3, acc[1][3]);
        __syncthreads();   // all waves done reading buf[cur] before it is DMA-overwritten
        cur ^= 1;
    }

#pragma unroll
    for (int mi = 0; mi < 2; ++mi) {
#pragma unroll
        for (int ni = 0; ni < 4; ++ni) {
            int col = colBlk + col0 + ni * 16 + ln;
            float bsv = bias[col];
#pragma unroll
            for (int r = 0; r < 8; ++r) {
                int row = rowBlk + row0 + mi * 16 + r + 8 * h;
                float v = acc[mi][ni][r] + bsv;
                if (op == 1) v = 0.5f * v * (1.0f + erff(v * 0.7071067811865476f));
                size_t o = (size_t)row * Nd_ + col;
                if (Cf)  Cf[o] = v;
                if (Cbf) Cbf[o] = f2bf(v);
            }
        }
    }
}

// ---------------- flash attention: one wave per (b, head, 32-query tile) ----------------
__global__ __launch_bounds__(32) void k_attn(const unsigned short* __restrict__ qbf,
                                             const unsigned short* __restrict__ kbf,
                                             const unsigned short* __restrict__ vtbf,
                                             const float4* __restrict__ coords,
                                             const float* __restrict__ extb,
                                             unsigned short* __restrict__ ctxbf) {
    const int lane = threadIdx.x;
    const int ln = lane & 15;
    const int h = (lane >> 4) & 1;
    const int q0 = blockIdx.x * 32;
    const int bh = blockIdx.y;
    const int b = bh / NHd;
    const int head = bh % NHd;
    const bool useX = head < (NHd / 2);

    __shared__ float4 qc_s[32];
    __shared__ unsigned short p_s[32 * 32];

    qc_s[lane] = coords[b * Sd + q0 + lane];

    const unsigned short* qb = qbf + (size_t)bh * Sd * DHd;
    const unsigned short* kb = kbf + (size_t)bh * Sd * DHd;
    const unsigned short* vb = vtbf + (size_t)bh * DHd * Sd;

    v16bf qa[2][2];
#pragma unroll
    for (int mi = 0; mi < 2; ++mi)
#pragma unroll
        for (int kc = 0; kc < 2; ++kc)
            qa[mi][kc] = frag_ld_a(qb, DHd, q0 + mi * 16, kc * 32, lane);

    const v8f zf = {0.f, 0.f, 0.f, 0.f, 0.f, 0.f, 0.f, 0.f};
    v8f O[2][4];
    float Mr[2][8], Lr[2][8], fac[2][8];
#pragma unroll
    for (int mi = 0; mi < 2; ++mi) {
#pragma unroll
        for (int ni = 0; ni < 4; ++ni) O[mi][ni] = zf;
#pragma unroll
        for (int r = 0; r < 8; ++r) { Mr[mi][r] = -1e30f; Lr[mi][r] = 0.0f; }
    }
    __syncthreads();

    for (int kt = 0; kt < Sd; kt += 32) {
        // ---- scores: Q (32x64) x K^T (64x32) ----
        v16bf kf[2][2];
#pragma unroll
        for (int ni = 0; ni < 2; ++ni)
#pragma unroll
            for (int kc = 0; kc < 2; ++kc)
                kf[ni][kc] = frag_ld_b(kb, DHd, kt + ni * 16, kc * 32, lane);

        v8f sf[2][2];
#pragma unroll
        for (int mi = 0; mi < 2; ++mi)
#pragma unroll
            for (int ni = 0; ni < 2; ++ni) {
                v8f t = wmma_bf16(qa[mi][0], kf[ni][0], zf);
                sf[mi][ni] = wmma_bf16(qa[mi][1], kf[ni][1], t);
            }

        // ---- bias: -local + ext_mask ----
        float4 kc0 = coords[b * Sd + kt + ln];
        float4 kc1 = coords[b * Sd + kt + 16 + ln];
        float e0 = extb[b * Sd + kt + ln];
        float e1 = extb[b * Sd + kt + 16 + ln];

#pragma unroll
        for (int mi = 0; mi < 2; ++mi) {
#pragma unroll
            for (int r = 0; r < 8; ++r) {
                float4 qc = qc_s[mi * 16 + r + 8 * h];
                float lo0, lo1;
                if (useX) {
                    lo0 = fminf(fminf(fabsf(qc.x - kc0.x), fabsf(qc.y - kc0.y)), fabsf(qc.z - kc0.z));
                    lo1 = fminf(fminf(fabsf(qc.x - kc1.x), fabsf(qc.y - kc1.y)), fabsf(qc.z - kc1.z));
                } else {
                    lo0 = fabsf(qc.w - kc0.w);
                    lo1 = fabsf(qc.w - kc1.w);
                }
                sf[mi][0][r] += e0 - lo0;
                sf[mi][1][r] += e1 - lo1;
            }
        }

        // ---- online softmax (rows live in 16-lane halves) ----
#pragma unroll
        for (int mi = 0; mi < 2; ++mi) {
#pragma unroll
            for (int r = 0; r < 8; ++r) {
                float t = fmaxf(sf[mi][0][r], sf[mi][1][r]);
                t = fmaxf(t, __shfl_xor(t, 1));
                t = fmaxf(t, __shfl_xor(t, 2));
                t = fmaxf(t, __shfl_xor(t, 4));
                t = fmaxf(t, __shfl_xor(t, 8));
                float Mn = fmaxf(Mr[mi][r], t);
                float fc = __expf(Mr[mi][r] - Mn);
                Mr[mi][r] = Mn;
                fac[mi][r] = fc;
                float p0 = __expf(sf[mi][0][r] - Mn);
                float p1 = __expf(sf[mi][1][r] - Mn);
                float rs = p0 + p1;
                rs += __shfl_xor(rs, 1);
                rs += __shfl_xor(rs, 2);
                rs += __shfl_xor(rs, 4);
                rs += __shfl_xor(rs, 8);
                Lr[mi][r] = Lr[mi][r] * fc + rs;
                int row = mi * 16 + r + 8 * h;
                p_s[row * 32 + ln] = f2bf(p0);
                p_s[row * 32 + 16 + ln] = f2bf(p1);
            }
#pragma unroll
            for (int ni = 0; ni < 4; ++ni)
#pragma unroll
                for (int r = 0; r < 8; ++r)
                    O[mi][ni][r] *= fac[mi][r];
        }
        __syncthreads();

        // ---- P (32x32) x V (32x64) ----
        v16bf pa0 = frag_ld_a(p_s, 32, 0, 0, lane);
        v16bf pa1 = frag_ld_a(p_s, 32, 16, 0, lane);
        v16bf vf[4];
#pragma unroll
        for (int ni = 0; ni < 4; ++ni)
            vf[ni] = frag_ld_b(vb, Sd, ni * 16, kt, lane);
#pragma unroll
        for (int ni = 0; ni < 4; ++ni) {
            O[0][ni] = wmma_bf16(pa0, vf[ni], O[0][ni]);
            O[1][ni] = wmma_bf16(pa1, vf[ni], O[1][ni]);
        }
        __syncthreads();
    }

    // ---- epilogue: ctx bf16 in [b, s, head, d] token-major layout ----
#pragma unroll
    for (int mi = 0; mi < 2; ++mi) {
#pragma unroll
        for (int r = 0; r < 8; ++r) {
            float inv = 1.0f / Lr[mi][r];
            int qrow = q0 + mi * 16 + r + 8 * h;
#pragma unroll
            for (int ni = 0; ni < 4; ++ni) {
                int d = ni * 16 + ln;
                ctxbf[((size_t)(b * Sd + qrow)) * Hd + head * DHd + d] = f2bf(O[mi][ni][r] * inv);
            }
        }
    }
}

// ---------------- residual + RMSNorm (reads y + h, writes h and bf16(h)) ----------------
__global__ __launch_bounds__(256) void k_rmsnorm(const float* __restrict__ y,
                                                 float* __restrict__ h,
                                                 const float* __restrict__ ns,
                                                 unsigned short* __restrict__ hbf) {
    int row = blockIdx.x;
    int t = threadIdx.x;
    __shared__ float red[256];
    float v[3];
    float ss = 0.0f;
#pragma unroll
    for (int i = 0; i < 3; ++i) {
        int idx = t + i * 256;
        float val = y[(size_t)row * Hd + idx] + h[(size_t)row * Hd + idx];
        v[i] = val;
        ss += val * val;
    }
    red[t] = ss;
    __syncthreads();
    for (int s = 128; s > 0; s >>= 1) {
        if (t < s) red[t] += red[t + s];
        __syncthreads();
    }
    float r = rsqrtf(red[0] / (float)Hd + 1e-5f);
#pragma unroll
    for (int i = 0; i < 3; ++i) {
        int idx = t + i * 256;
        float o = ns[idx] * v[i] * r;
        h[(size_t)row * Hd + idx] = o;
        hbf[(size_t)row * Hd + idx] = f2bf(o);
    }
}

// ---------------- host launch ----------------
extern "C" void kernel_launch(void* const* d_in, const int* in_sizes, int n_in,
                              void* d_out, int out_size, void* d_ws, size_t ws_size,
                              hipStream_t stream) {
    const float* hidden = (const float*)d_in[0];
    const int* bbox = (const int*)d_in[1];
    const int* amask = (const int*)d_in[2];
    const float* Wq = (const float*)d_in[3];
    const float* bq = (const float*)d_in[4];
    const float* Wk = (const float*)d_in[5];
    const float* bk = (const float*)d_in[6];
    const float* Wv = (const float*)d_in[7];
    const float* bv = (const float*)d_in[8];
    const float* Wo = (const float*)d_in[9];
    const float* bo = (const float*)d_in[10];
    const float* ns1 = (const float*)d_in[11];
    const float* W1 = (const float*)d_in[12];
    const float* b1 = (const float*)d_in[13];
    const float* W2 = (const float*)d_in[14];
    const float* b2 = (const float*)d_in[15];
    const float* ns2 = (const float*)d_in[16];
    float* hbuf = (float*)d_out; // running hidden state lives in d_out

    // workspace carve-up
    unsigned char* w = (unsigned char*)d_ws;
    size_t off = 0;
    auto nxt = [&](size_t bytes) -> void* {
        void* p = w + off;
        off += (bytes + 255) & ~(size_t)255;
        return p;
    };
    unsigned short* WqT = (unsigned short*)nxt((size_t)LAY * Hd * Hd * 2);
    unsigned short* WkT = (unsigned short*)nxt((size_t)LAY * Hd * Hd * 2);
    unsigned short* WvT = (unsigned short*)nxt((size_t)LAY * Hd * Hd * 2);
    unsigned short* WoT = (unsigned short*)nxt((size_t)LAY * Hd * Hd * 2);
    unsigned short* W1T = (unsigned short*)nxt((size_t)LAY * Hd * FFd * 2);
    unsigned short* W2T = (unsigned short*)nxt((size_t)LAY * FFd * Hd * 2);
    unsigned short* hbf = (unsigned short*)nxt((size_t)Md * Hd * 2);
    float* qC = (float*)nxt((size_t)Md * Hd * 4);
    float* kC = (float*)nxt((size_t)Md * Hd * 4);
    float* vC = (float*)nxt((size_t)Md * Hd * 4);
    unsigned short* qbf = (unsigned short*)nxt((size_t)Md * Hd * 2);
    unsigned short* kbf = (unsigned short*)nxt((size_t)Md * Hd * 2);
    unsigned short* vtbf = (unsigned short*)nxt((size_t)Md * Hd * 2);
    unsigned short* ctxbf = (unsigned short*)nxt((size_t)Md * Hd * 2);
    unsigned short* ffbf = (unsigned short*)nxt((size_t)Md * FFd * 2);
    float* s1 = (float*)nxt((size_t)Bd * Sd * 32 * 4);
    float* c1 = (float*)nxt((size_t)Bd * Sd * 32 * 4);
    float* s2 = (float*)nxt((size_t)Bd * Sd * 32 * 4);
    float* c2 = (float*)nxt((size_t)Bd * Sd * 32 * 4);
    float4* coords = (float4*)nxt((size_t)Bd * Sd * sizeof(float4));
    float* extb = (float*)nxt((size_t)Bd * Sd * 4);
    (void)ws_size; (void)in_sizes; (void)n_in; (void)out_size;

    // ---- prep ----
    {
        int tot = LAY * Hd * Hd;
        k_w_transpose<<<(tot + 255) / 256, 256, 0, stream>>>(Wq, WqT, Hd, Hd, tot);
        k_w_transpose<<<(tot + 255) / 256, 256, 0, stream>>>(Wk, WkT, Hd, Hd, tot);
        k_w_transpose<<<(tot + 255) / 256, 256, 0, stream>>>(Wv, WvT, Hd, Hd, tot);
        k_w_transpose<<<(tot + 255) / 256, 256, 0, stream>>>(Wo, WoT, Hd, Hd, tot);
        int tot1 = LAY * Hd * FFd;
        k_w_transpose<<<(tot1 + 255) / 256, 256, 0, stream>>>(W1, W1T, Hd, FFd, tot1);
        k_w_transpose<<<(tot1 + 255) / 256, 256, 0, stream>>>(W2, W2T, FFd, Hd, tot1);
    }
    k_tables<<<(Bd * Sd + 255) / 256, 256, 0, stream>>>(bbox, amask, s1, c1, s2, c2, coords, extb);
    {
        int tot = Md * Hd;
        k_init<<<(tot + 255) / 256, 256, 0, stream>>>(hidden, hbuf, hbf, tot);
    }

    const dim3 gemmBlk(256);
    const int ropeTot = Bd * Sd * NHd * 32;
    const int vreTot = Bd * NHd * DHd * Sd;

    for (int l = 0; l < LAY; ++l) {
        const unsigned short* WqTl = WqT + (size_t)l * Hd * Hd;
        const unsigned short* WkTl = WkT + (size_t)l * Hd * Hd;
        const unsigned short* WvTl = WvT + (size_t)l * Hd * Hd;
        const unsigned short* WoTl = WoT + (size_t)l * Hd * Hd;
        const unsigned short* W1Tl = W1T + (size_t)l * Hd * FFd;
        const unsigned short* W2Tl = W2T + (size_t)l * FFd * Hd;

        // QKV projections
        k_gemm_bf16<<<dim3(Md / 128, Hd / 128), gemmBlk, 0, stream>>>(hbf, WqTl, bq + l * Hd, qC, nullptr, Hd, Hd, 0);
        k_gemm_bf16<<<dim3(Md / 128, Hd / 128), gemmBlk, 0, stream>>>(hbf, WkTl, bk + l * Hd, kC, nullptr, Hd, Hd, 0);
        k_gemm_bf16<<<dim3(Md / 128, Hd / 128), gemmBlk, 0, stream>>>(hbf, WvTl, bv + l * Hd, vC, nullptr, Hd, Hd, 0);

        // RoPE (scale folded into Q) + V transpose
        k_rope<<<(ropeTot + 255) / 256, 256, 0, stream>>>(qC, qbf, s1, c1, s2, c2, 0.125f);
        k_rope<<<(ropeTot + 255) / 256, 256, 0, stream>>>(kC, kbf, s1, c1, s2, c2, 1.0f);
        k_vre<<<(vreTot + 255) / 256, 256, 0, stream>>>(vC, vtbf);

        // attention
        k_attn<<<dim3(Sd / 32, Bd * NHd), 32, 0, stream>>>(qbf, kbf, vtbf, coords, extb, ctxbf);

        // output projection + residual RMSNorm 1
        k_gemm_bf16<<<dim3(Md / 128, Hd / 128), gemmBlk, 0, stream>>>(ctxbf, WoTl, bo + l * Hd, qC, nullptr, Hd, Hd, 0);
        k_rmsnorm<<<Md, 256, 0, stream>>>(qC, hbuf, ns1 + l * Hd, hbf);

        // FFN: GELU(h @ W1 + b1) @ W2 + b2, residual RMSNorm 2
        k_gemm_bf16<<<dim3(Md / 128, FFd / 128), gemmBlk, 0, stream>>>(hbf, W1Tl, b1 + l * FFd, nullptr, ffbf, FFd, Hd, 1);
        k_gemm_bf16<<<dim3(Md / 128, Hd / 128), gemmBlk, 0, stream>>>(ffbf, W2Tl, b2 + l * Hd, kC, nullptr, Hd, FFd, 0);
        k_rmsnorm<<<Md, 256, 0, stream>>>(kC, hbuf, ns2 + l * Hd, hbf);
    }
}
